// KANLayer_75041668596051
// MI455X (gfx1250) — compile-verified, matrix operand order checked by reading, not run on hardware
//
#include <hip/hip_runtime.h>
#include <hip/hip_bf16.h>

#define DIM        1024
#define N_TOKENS   8192
#define GRID_SIZE  5
#define EPS        1e-6f

typedef __attribute__((ext_vector_type(16))) _Float16 v16h;
typedef __attribute__((ext_vector_type(8)))  _Float16 v8h;
typedef __attribute__((ext_vector_type(4)))  _Float16 v4h;
typedef __attribute__((ext_vector_type(8)))  float    v8f;

// ---------------------------------------------------------------------------
// Cubic B-spline (Cox-de Boor) on uniform knots t_i = -2.2 + 0.55*i, i=0..8.
// All denominators in the reference reduce to k*h exactly (h = 4.4/8 = 0.55).
// ---------------------------------------------------------------------------
__device__ __forceinline__ float spline_eval(float x, const float* __restrict__ w) {
    const float H = 0.55f;
    float t[9];
#pragma unroll
    for (int i = 0; i < 9; ++i) t[i] = -2.2f + H * (float)i;

    float b0[8];
#pragma unroll
    for (int i = 0; i < 8; ++i)
        b0[i] = (x >= t[i] && x < t[i + 1]) ? 1.0f : 0.0f;

    const float ih1 = 1.0f / H;
    float b1[7];
#pragma unroll
    for (int i = 0; i < 7; ++i)
        b1[i] = (x - t[i]) * ih1 * b0[i] + (t[i + 2] - x) * ih1 * b0[i + 1];

    const float ih2 = 1.0f / (2.0f * H);
    float b2[6];
#pragma unroll
    for (int i = 0; i < 6; ++i)
        b2[i] = (x - t[i]) * ih2 * b1[i] + (t[i + 3] - x) * ih2 * b1[i + 1];

    const float ih3 = 1.0f / (3.0f * H);
    float b3[5];
#pragma unroll
    for (int i = 0; i < 5; ++i)
        b3[i] = (x - t[i]) * ih3 * b2[i] + (t[i + 4] - x) * ih3 * b2[i + 1];

    float acc = 0.0f;
#pragma unroll
    for (int i = 0; i < 5; ++i) acc = fmaf(b3[i], w[i], acc);
    return acc;
}

// ---------------------------------------------------------------------------
// Kernel 1: per-row RMSNorm, emit f16 x_norm for the WMMA GEMM, and seed
// out = x + spline_out * gamma. One 256-thread block (8 waves) per row.
// ---------------------------------------------------------------------------
__global__ void __launch_bounds__(256)
kan_norm_spline(const float* __restrict__ x,
                const float* __restrict__ norm_w,
                const float* __restrict__ spline_w,
                const float* __restrict__ gamma,
                _Float16* __restrict__ xn16,
                float* __restrict__ out)
{
    const int row  = blockIdx.x;
    const int tid  = threadIdx.x;
    const float4* xr4 = (const float4*)(x + (size_t)row * DIM);
    float4 v = xr4[tid];

    float ss = v.x * v.x + v.y * v.y + v.z * v.z + v.w * v.w;
#pragma unroll
    for (int off = 16; off > 0; off >>= 1)
        ss += __shfl_down(ss, off, 32);

    __shared__ float red[8];
    const int wave = tid >> 5, lane = tid & 31;
    if (lane == 0) red[wave] = ss;
    __syncthreads();
    if (tid == 0) {
        float t = 0.0f;
#pragma unroll
        for (int i = 0; i < 8; ++i) t += red[i];
        red[0] = rsqrtf(t * (1.0f / (float)DIM) + EPS);
    }
    __syncthreads();
    const float r = red[0];

    const int d0 = tid * 4;
    float xs[4] = {v.x, v.y, v.z, v.w};
    _Float16 h4[4];
    float    o4[4];
#pragma unroll
    for (int c = 0; c < 4; ++c) {
        const int d = d0 + c;
        const float xn = xs[c] * r * norm_w[d];
        h4[c] = (_Float16)xn;
        const float xc = fminf(fmaxf(xn, -1.0f), 1.0f);
        const float sp = spline_eval(xc, spline_w + (size_t)d * GRID_SIZE);
        o4[c] = xs[c] + sp * gamma[d];
    }
    v4h hv = {h4[0], h4[1], h4[2], h4[3]};
    *(v4h*)(xn16 + (size_t)row * DIM + d0) = hv;
    float4 ov = {o4[0], o4[1], o4[2], o4[3]};
    *(float4*)(out + (size_t)row * DIM + d0) = ov;
}

// ---------------------------------------------------------------------------
// Kernel 2: fp32 -> f16 weight conversion (W stays L2-resident afterwards).
// ---------------------------------------------------------------------------
__global__ void __launch_bounds__(256)
kan_convert_w(const float4* __restrict__ w, v4h* __restrict__ wh, int n4)
{
    const int i = blockIdx.x * blockDim.x + threadIdx.x;
    if (i < n4) {
        float4 f = w[i];
        v4h h = {(_Float16)f.x, (_Float16)f.y, (_Float16)f.z, (_Float16)f.w};
        wh[i] = h;
    }
}

// ---------------------------------------------------------------------------
// Async-to-LDS 16-byte copy (gfx1250 GLOBAL_LOAD_ASYNC_TO_LDS_B128, ASYNCcnt).
// Generic pointer to __shared__ carries the LDS offset in its low 32 bits.
// ---------------------------------------------------------------------------
__device__ __forceinline__ void async_copy16(const _Float16* g, _Float16* l) {
    unsigned int loff = (unsigned int)(uintptr_t)l;
    asm volatile("global_load_async_to_lds_b128 %0, %1, off"
                 :: "v"(loff), "v"(g) : "memory");
}

__device__ __forceinline__ void wait_async0() {
    asm volatile("s_wait_asynccnt 0x0" ::: "memory");
}

// ---------------------------------------------------------------------------
// Kernel 3: base_out = x_norm @ W^T via v_wmma_f32_16x16x32_f16 with
// double-buffered async LDS staging; fused epilogue out += base_out * gamma.
//
// Block = 8 waves = 128(M) x 64(N) tile. Per K-step of 64 halves:
//   A stage 128x72 halves (pad 8 -> 144B row stride, conflict-free b128 reads)
//   B stage  64x72 halves (B shared by all 8 waves: 8x fewer B fetches)
// Double buffered: fill(k+1) overlaps compute(k); sync = s_wait_asynccnt 0
// + s_barrier. Each wave: 16x64 output (4 f32 accumulators).
// ---------------------------------------------------------------------------
#define KSTEP  64
#define ARS    72                  // tile row stride in halves (64 + 8 pad)
#define A_TILE (128 * ARS)         // 9216 halves
#define B_TILE (64 * ARS)          // 4608 halves
#define BUFSZ  (A_TILE + B_TILE)   // 13824 halves = 27 KB per buffer

__device__ __forceinline__ void fill_tile(const _Float16* __restrict__ A,
                                          const _Float16* __restrict__ W,
                                          _Float16* __restrict__ buf,
                                          int mblock, int n0, int k, int tid)
{
    // A tile: 128 rows x 64 halves = 1024 16B chunks, 4 per thread
#pragma unroll
    for (int t = 0; t < 4; ++t) {
        const int c   = tid + t * 256;
        const int row = c >> 3;            // 8 chunks per row
        const int col = (c & 7) * 8;       // half index within row
        async_copy16(A + (size_t)(mblock + row) * DIM + k + col,
                     buf + row * ARS + col);
    }
    // B tile: 64 rows x 64 halves = 512 chunks, 2 per thread
#pragma unroll
    for (int t = 0; t < 2; ++t) {
        const int c   = tid + t * 256;
        const int row = c >> 3;
        const int col = (c & 7) * 8;
        async_copy16(W + (size_t)(n0 + row) * DIM + k + col,
                     buf + A_TILE + row * ARS + col);
    }
}

#define LDS_FRAG(base, off)                                                   \
    __builtin_shufflevector(*(const v8h*)((base) + (off)),                    \
                            *(const v8h*)((base) + (off) + 16),               \
                            0, 1, 2, 3, 4, 5, 6, 7, 8, 9, 10, 11, 12, 13, 14, 15)

__global__ void __launch_bounds__(256)
kan_gemm(const _Float16* __restrict__ A,   // N_TOKENS x DIM (f16 x_norm)
         const _Float16* __restrict__ W,   // DIM x DIM (f16, row = out feature)
         const float* __restrict__ gamma,
         float* __restrict__ out)          // accumulate base*gamma
{
    __shared__ _Float16 smem[2 * BUFSZ];

    const int tid    = threadIdx.x;
    const int lane   = tid & 31;
    const int wave   = tid >> 5;
    const int mblock = blockIdx.y * 128;
    const int n0     = blockIdx.x * 64;
    const int r      = lane & 15;
    const int koff   = (lane >> 4) << 3;   // 0 or 8

    v8f acc[4] = {v8f{}, v8f{}, v8f{}, v8f{}};

    fill_tile(A, W, &smem[0], mblock, n0, 0, tid);

    const int NT = DIM / KSTEP;            // 16 K-steps
    for (int kt = 0; kt < NT; ++kt) {
        _Float16* cur = &smem[(kt & 1) * BUFSZ];

        wait_async0();        // this wave's fill of `cur` landed in LDS
        __syncthreads();      // everyone's fill landed

        if (kt + 1 < NT)      // overlap: stream next K-step into other buffer
            fill_tile(A, W, &smem[((kt + 1) & 1) * BUFSZ],
                      mblock, n0, (kt + 1) * KSTEP, tid);

        const _Float16* aB = cur + (wave * 16 + r) * ARS + koff;
        const _Float16* bB = cur + A_TILE + r * ARS + koff;
#pragma unroll
        for (int ks = 0; ks < KSTEP; ks += 32) {
            v16h fa = LDS_FRAG(aB, ks);
            v16h f0 = LDS_FRAG(bB + 0 * 16 * ARS, ks);
            v16h f1 = LDS_FRAG(bB + 1 * 16 * ARS, ks);
            v16h f2 = LDS_FRAG(bB + 2 * 16 * ARS, ks);
            v16h f3 = LDS_FRAG(bB + 3 * 16 * ARS, ks);

            acc[0] = __builtin_amdgcn_wmma_f32_16x16x32_f16(false, fa, false, f0,
                                                            (short)0, acc[0], false, false);
            acc[1] = __builtin_amdgcn_wmma_f32_16x16x32_f16(false, fa, false, f1,
                                                            (short)0, acc[1], false, false);
            acc[2] = __builtin_amdgcn_wmma_f32_16x16x32_f16(false, fa, false, f2,
                                                            (short)0, acc[2], false, false);
            acc[3] = __builtin_amdgcn_wmma_f32_16x16x32_f16(false, fa, false, f3,
                                                            (short)0, acc[3], false, false);
        }
        __syncthreads();      // all waves done reading `cur` before refill
    }

    // epilogue: out[m,n] += acc * gamma[n]   (C/D layout per ISA 7.12.2)
    const int m0    = mblock + wave * 16;
    const int mbase = m0 + ((lane >> 4) << 3);
#pragma unroll
    for (int j = 0; j < 4; ++j) {
        const int nn = n0 + j * 16 + (lane & 15);
        const float g = gamma[nn];
        float* op = out + (size_t)mbase * DIM + nn;
#pragma unroll
        for (int vv = 0; vv < 8; ++vv)
            op[(size_t)vv * DIM] += acc[j][vv] * g;
    }
}

// ---------------------------------------------------------------------------
// Launch: inputs are (x, norm_weight, base_weight, spline_weight, gamma).
// d_ws: [0, 16MB) f16 x_norm, [16MB, 18MB) f16 weights.
// ---------------------------------------------------------------------------
extern "C" void kernel_launch(void* const* d_in, const int* in_sizes, int n_in,
                              void* d_out, int out_size, void* d_ws, size_t ws_size,
                              hipStream_t stream) {
    const float* x        = (const float*)d_in[0];
    const float* norm_w   = (const float*)d_in[1];
    const float* base_w   = (const float*)d_in[2];
    const float* spline_w = (const float*)d_in[3];
    const float* gamma    = (const float*)d_in[4];
    float* out = (float*)d_out;

    _Float16* xn16 = (_Float16*)d_ws;
    _Float16* wh   = xn16 + (size_t)N_TOKENS * DIM;

    kan_norm_spline<<<N_TOKENS, 256, 0, stream>>>(x, norm_w, spline_w, gamma, xn16, out);

    const int n4 = (DIM * DIM) / 4;
    kan_convert_w<<<(n4 + 255) / 256, 256, 0, stream>>>((const float4*)base_w, (v4h*)wh, n4);

    dim3 grid(DIM / 64, N_TOKENS / 128);
    kan_gemm<<<grid, 256, 0, stream>>>(xn16, wh, gamma, out);
}
